// AddWithCarryNetwork_7645041787323
// MI455X (gfx1250) — matrix operand order repaired and checked by reading, not compiled, not run on hardware
//
#include <hip/hip_runtime.h>
#include <stdint.h>

typedef __attribute__((ext_vector_type(16))) _Float16 v16h;
typedef __attribute__((ext_vector_type(8)))  float    v8f;

#define BSZ    2048
#define NBITS  4096
#define WPB    4       // waves per block
#define TILE   32      // bit-columns staged per chunk
#define XPITCH 36      // dwords per staged row: 16B-aligned rows, conflict-free reads

__device__ __forceinline__ float fast_sigmoid(float z) {
#if __has_builtin(__builtin_amdgcn_tanhf)
    // sigmoid(z) = 0.5*tanh(z/2) + 0.5  -> single v_tanh_f32 transcendental
    return fmaf(0.5f, __builtin_amdgcn_tanhf(0.5f * z), 0.5f);
#else
    float e = __expf(-z);
    return __builtin_amdgcn_rcpf(1.0f + e);
#endif
}

__global__ void __launch_bounds__(WPB * 32, 1)
addcarry_wmma_kernel(const float* __restrict__ xg, const float* __restrict__ yg,
                     const float* __restrict__ W1, const float* __restrict__ b1,
                     const float* __restrict__ W2, const float* __restrict__ b2,
                     float* __restrict__ outg)
{
    // Per-wave LDS slices (no inter-wave sharing -> no barriers anywhere).
    __shared__ alignas(16) float xt[WPB][16 * XPITCH];   // x tile, [row][col-asc]
    __shared__ alignas(16) float yt[WPB][16 * XPITCH];   // y tile
    __shared__ alignas(16) float st[WPB][2][TILE * 16];  // sum bits, [step][row], dbl-buffered
    __shared__ alignas(16) float carr[WPB][16];          // per-row carry broadcast slot

    const int lane = threadIdx.x & 31;
    const int w    = threadIdx.x >> 5;
    const int n    = lane & 15;          // WMMA D column owned
    const int r    = n;                  // batch row (in tile) this lane builds A for
    const int hihalf = lane >> 4;
    const int klo  = hihalf ? 8 : 0;     // A-fragment K ownership (f16 A 16x32 layout)
    const int kb   = hihalf ? 16 : 0;    // unused: B fragment takes K from klo, same order as A
    const int m0   = hihalf ? 8 : 0;     // D rows held by this lane group
    const long row0 = ((long)blockIdx.x * WPB + w) * 16;

    // ---- layer-1 weights, per-lane slice matching the A-fragment K layout ----
    float wx[16], wy[16], wc[16], bb[16];
    #pragma unroll
    for (int u = 0; u < 16; ++u) {
        int j = klo + (u & 7) + ((u >> 3) << 4);
        wx[u] = W1[     j];
        wy[u] = W1[32 + j];
        wc[u] = W1[64 + j];
        bb[u] = b1[j];
    }

    // ---- layer-2 B fragment: W2 (32x2) zero-padded to 32x16, f16 ----
    v16h bf;
    #pragma unroll
    for (int t = 0; t < 16; ++t) {
        float wv = (n < 2) ? W2[(klo + (t & 7) + ((t >> 3) << 4)) * 2 + n] : 0.0f;
        bf[t] = (_Float16)wv;
    }
    // ---- bias b2 as C fragment ----
    v8f cf;
    {
        float c0 = (n < 2) ? b2[n] : 0.0f;
        #pragma unroll
        for (int v = 0; v < 8; ++v) cf[v] = c0;
    }

    const int lrr = lane >> 3;           // DMA lane mapping: 4 rows x 8 lanes x 16B
    const int lc4 = (lane & 7) * 4;

    float cv = 0.0f;                     // carry (LSB-first scan starts at 0)

    for (int tb = 0; tb < NBITS; tb += TILE) {
        const int gbase = NBITS - TILE - tb;   // tile col c <-> global col gbase+c; step s -> c=31-s

        // ---- stage x/y tile: 4x b128 async DMA per operand, 128B/row coalesced ----
        #pragma unroll
        for (int g4 = 0; g4 < 4; ++g4) {
            int rr = g4 * 4 + lrr;
            uint64_t gx = (uint64_t)(uintptr_t)(const void*)(xg + (row0 + rr) * (long)NBITS + gbase + lc4);
            uint64_t gy = (uint64_t)(uintptr_t)(const void*)(yg + (row0 + rr) * (long)NBITS + gbase + lc4);
            uint32_t lx = (uint32_t)(uintptr_t)(void*)&xt[w][rr * XPITCH + lc4];
            uint32_t ly = (uint32_t)(uintptr_t)(void*)&yt[w][rr * XPITCH + lc4];
            asm volatile("global_load_async_to_lds_b128 %0, %1, off" :: "v"(lx), "v"(gx) : "memory");
            asm volatile("global_load_async_to_lds_b128 %0, %1, off" :: "v"(ly), "v"(gy) : "memory");
        }
        asm volatile("s_wait_asynccnt 0x0" ::: "memory");  // tile in LDS; also drains old flushes

        const int cb = (tb >> 5) & 1;

        // ---- prime software pipeline: carry-independent layer-1 partial for s=0 ----
        float xv = xt[w][r * XPITCH + 31];
        float yv = yt[w][r * XPITCH + 31];
        float hb[16];
        #pragma unroll
        for (int u = 0; u < 16; ++u)
            hb[u] = fmaf(wy[u], yv, fmaf(wx[u], xv, bb[u]));

        for (int s = 0; s < TILE; ++s) {
            // ---- finish layer 1 (only the carry term is on the serial chain) ----
            v16h af;
            #pragma unroll
            for (int u = 0; u < 16; ++u)
                af[u] = (_Float16)fast_sigmoid(fmaf(wc[u], cv, hb[u]));

            // ---- layer 2: D = h x W2pad + b2 ----
            v8f acc = __builtin_amdgcn_wmma_f32_16x16x32_f16(
                false, af, false, bf, (short)0, cf, false, false);

            // ---- next step's carry-independent partial: runs in the WMMA shadow ----
            if (s < TILE - 1) {
                xv = xt[w][r * XPITCH + 30 - s];
                yv = yt[w][r * XPITCH + 30 - s];
            }
            #pragma unroll
            for (int u = 0; u < 16; ++u)
                hb[u] = fmaf(wy[u], yv, fmaf(wx[u], xv, bb[u]));

            // ---- lanes owning D columns 0 (sum) / 1 (carry) post-process ----
            if (n < 2) {
                float sg[8];
                #pragma unroll
                for (int k2 = 0; k2 < 8; ++k2) sg[k2] = fast_sigmoid(acc[k2]);
                if (n == 0) {
                    float4* p = (float4*)&st[w][cb][s * 16 + m0];
                    p[0] = make_float4(sg[0], sg[1], sg[2], sg[3]);
                    p[1] = make_float4(sg[4], sg[5], sg[6], sg[7]);
                } else {
                    float4* p = (float4*)&carr[w][m0];
                    p[0] = make_float4(sg[0], sg[1], sg[2], sg[3]);
                    p[1] = make_float4(sg[4], sg[5], sg[6], sg[7]);
                }
            }
            asm volatile("s_wait_dscnt 0x0" ::: "memory");
            cv = carr[w][r];
        }

        // ---- flush sum-bit tile: async DMA LDS -> global (coalesced per row) ----
        #pragma unroll 4
        for (int rr = 0; rr < 16; ++rr) {
            uint64_t go = (uint64_t)(uintptr_t)(void*)(outg + (row0 + rr) * (long)NBITS + gbase + lane);
            uint32_t lo = (uint32_t)(uintptr_t)(void*)&st[w][cb][(31 - lane) * 16 + rr];
            asm volatile("global_store_async_from_lds_b32 %0, %1, off" :: "v"(go), "v"(lo) : "memory");
        }
    }
    asm volatile("s_wait_asynccnt 0x0" ::: "memory");  // drain final flush
}

extern "C" void kernel_launch(void* const* d_in, const int* in_sizes, int n_in,
                              void* d_out, int out_size, void* d_ws, size_t ws_size,
                              hipStream_t stream) {
    const float* x  = (const float*)d_in[0];
    const float* y  = (const float*)d_in[1];
    const float* W1 = (const float*)d_in[2];
    const float* b1 = (const float*)d_in[3];
    const float* W2 = (const float*)d_in[4];
    const float* b2 = (const float*)d_in[5];
    float* out = (float*)d_out;

    dim3 grid(BSZ / (16 * WPB));   // 32 blocks
    dim3 block(WPB * 32);          // 4 wave32 waves per block -> 128 wave-chains
    hipLaunchKernelGGL(addcarry_wmma_kernel, grid, block, 0, stream,
                       x, y, W1, b1, W2, b2, out);
}
